// S2ConvNet_deep_var_anchored_58884001628195
// MI455X (gfx1250) — compile-verified, matrix operand order checked by reading, not run on hardware
//
#include <hip/hip_runtime.h>
#include <hip/hip_bf16.h>

#define PI_D 3.14159265358979323846

typedef __attribute__((ext_vector_type(16))) _Float16 v16h;
typedef __attribute__((ext_vector_type(8)))  float    v8f;

namespace {

// ---------------------------------------------------------------------------
// Fast WMMA GEMM: C[z,r,n] = op( sum_k A[z,r,k] * Bt[n,k] )
//  - A: fp32, float-contiguous along k (row stride sam, z-slab stride saz).
//    Interleaved complex64 activations are "float contiguous" views.
//  - Bt: fp32 row-major TRANSPOSED weights [Nf][ldbt] (device-generated:
//    DFT matrices / kernel FTs in 2x2 real-block complex expansion / MLP W^T).
//  - C: scattered store with split (o,n) output axes, optional bias+ReLU.
//  One wave = 32x16 output tile (2 WMMAs share the B fragment).
//  f16 inputs, f32 accumulation: v_wmma_f32_16x16x32_f16 (wave32).
// ---------------------------------------------------------------------------
struct GemmFArgs {
  const float* A;
  long long saz, sam;
  int M, Kf;
  const float* Bt;
  int ldbt, Nf;
  float* C;
  long long scz, scm, sco, scn;
  int ndiv, cCplx;
  const float* bias;
  int relu;
};

__global__ __launch_bounds__(32) void k_gemmf(GemmFArgs g) {
  const int lane = threadIdx.x;
  const int row = lane & 15, grp = lane >> 4;
  const int tm = blockIdx.x << 5;
  const int tn = blockIdx.y << 4;
  const int r0 = tm + row, r1 = tm + 16 + row;
  const int n = tn + row;
  const long long zA = (long long)blockIdx.z * g.saz;
  // clamped bases: rows/cols >= bounds load valid-but-unused data; their
  // accumulators are never stored, so no exec-mask churn in the hot loop.
  const int rc0 = r0 < g.M ? r0 : g.M - 1;
  const int rc1 = r1 < g.M ? r1 : g.M - 1;
  const int nc = n < g.Nf ? n : g.Nf - 1;
  const float* A0 = g.A + zA + (long long)rc0 * g.sam + (grp << 3);
  const float* A1 = g.A + zA + (long long)rc1 * g.sam + (grp << 3);
  const float* Bp = g.Bt + (long long)nc * g.ldbt + (grp << 3);
  v8f acc0 = {}, acc1 = {};
  const int kmain = g.Kf & ~31;
  for (int k0 = 0; k0 < kmain; k0 += 32) {
    // Issue all loads first (one VMEM clause, staged waits), then convert.
    // CDNA5 16-bit fragment striping: lane-group k-ranges {0-7,16-23} /
    // {8-15,24-31}; each is two contiguous 8-float runs.
    float2 u0[4], u1[4], u2[4], u3[4], u4[4], u5[4];
#pragma unroll
    for (int t = 0; t < 4; t++) {
      u0[t] = *(const float2*)(A0 + k0 + 2 * t);
      u1[t] = *(const float2*)(A0 + k0 + 16 + 2 * t);
      u2[t] = *(const float2*)(A1 + k0 + 2 * t);
      u3[t] = *(const float2*)(A1 + k0 + 16 + 2 * t);
      u4[t] = *(const float2*)(Bp + k0 + 2 * t);
      u5[t] = *(const float2*)(Bp + k0 + 16 + 2 * t);
    }
    __builtin_prefetch(A0 + k0 + 128, 0, 3);   // global_prefetch_b8 (near)
    __builtin_prefetch(Bp + k0 + 128, 0, 3);
    v16h a0, a1, b;
#pragma unroll
    for (int t = 0; t < 4; t++) {
      a0[2*t] = (_Float16)u0[t].x; a0[2*t+1] = (_Float16)u0[t].y;
      a0[8+2*t] = (_Float16)u1[t].x; a0[9+2*t] = (_Float16)u1[t].y;
      a1[2*t] = (_Float16)u2[t].x; a1[2*t+1] = (_Float16)u2[t].y;
      a1[8+2*t] = (_Float16)u3[t].x; a1[9+2*t] = (_Float16)u3[t].y;
      b[2*t] = (_Float16)u4[t].x; b[2*t+1] = (_Float16)u4[t].y;
      b[8+2*t] = (_Float16)u5[t].x; b[9+2*t] = (_Float16)u5[t].y;
    }
    acc0 = __builtin_amdgcn_wmma_f32_16x16x32_f16(false, a0, false, b, (short)0, acc0, false, false);
    acc1 = __builtin_amdgcn_wmma_f32_16x16x32_f16(false, a1, false, b, (short)0, acc1, false, false);
  }
  if (kmain < g.Kf) {
    // Branch-free K tail, same load shape as the main loop: unconditional
    // contiguous slab loads (<=124B overread stays inside the ws arena /
    // input allocation page), then mask the converted values via cndmask.
    float2 u0[4], u1[4], u2[4], u3[4], u4[4], u5[4];
#pragma unroll
    for (int t = 0; t < 4; t++) {
      u0[t] = *(const float2*)(A0 + kmain + 2 * t);
      u1[t] = *(const float2*)(A0 + kmain + 16 + 2 * t);
      u2[t] = *(const float2*)(A1 + kmain + 2 * t);
      u3[t] = *(const float2*)(A1 + kmain + 16 + 2 * t);
      u4[t] = *(const float2*)(Bp + kmain + 2 * t);
      u5[t] = *(const float2*)(Bp + kmain + 16 + 2 * t);
    }
    const int kbase = kmain + (grp << 3);
    v16h a0, a1, b;
#pragma unroll
    for (int t = 0; t < 4; t++) {
      const bool o0 = (kbase + 2 * t) < g.Kf,      o1 = (kbase + 2 * t + 1) < g.Kf;
      const bool o2 = (kbase + 16 + 2 * t) < g.Kf, o3 = (kbase + 16 + 2 * t + 1) < g.Kf;
      a0[2*t] = (_Float16)(o0 ? u0[t].x : 0.f); a0[2*t+1] = (_Float16)(o1 ? u0[t].y : 0.f);
      a0[8+2*t] = (_Float16)(o2 ? u1[t].x : 0.f); a0[9+2*t] = (_Float16)(o3 ? u1[t].y : 0.f);
      a1[2*t] = (_Float16)(o0 ? u2[t].x : 0.f); a1[2*t+1] = (_Float16)(o1 ? u2[t].y : 0.f);
      a1[8+2*t] = (_Float16)(o2 ? u3[t].x : 0.f); a1[9+2*t] = (_Float16)(o3 ? u3[t].y : 0.f);
      b[2*t] = (_Float16)(o0 ? u4[t].x : 0.f); b[2*t+1] = (_Float16)(o1 ? u4[t].y : 0.f);
      b[8+2*t] = (_Float16)(o2 ? u5[t].x : 0.f); b[9+2*t] = (_Float16)(o3 ? u5[t].y : 0.f);
    }
    acc0 = __builtin_amdgcn_wmma_f32_16x16x32_f16(false, a0, false, b, (short)0, acc0, false, false);
    acc1 = __builtin_amdgcn_wmma_f32_16x16x32_f16(false, a1, false, b, (short)0, acc1, false, false);
  }
  float* Cz = g.C + (long long)blockIdx.z * g.scz;
  if (n < g.Nf) {
    const float bval = g.bias ? g.bias[n] : 0.f;
    long long cad;
    if (g.cCplx) {
      const int cc = n >> 1;
      cad = (long long)(cc / g.ndiv) * g.sco + (long long)(cc % g.ndiv) * g.scn + (n & 1);
    } else {
      cad = (long long)(n / g.ndiv) * g.sco + (long long)(n % g.ndiv) * g.scn;
    }
#pragma unroll
    for (int v = 0; v < 8; v++) {
      const int rr0 = tm + v + (grp << 3);  // C layout: VGPR v -> M = v + 8*grp
      const int rr1 = rr0 + 16;
      if (rr0 < g.M) {
        float val = acc0[v] + bval; if (g.relu) val = fmaxf(val, 0.f);
        Cz[(long long)rr0 * g.scm + cad] = val;
      }
      if (rr1 < g.M) {
        float val = acc1[v] + bval; if (g.relu) val = fmaxf(val, 0.f);
        Cz[(long long)rr1 * g.scm + cad] = val;
      }
    }
  }
}

// --------------------------- precompute kernels ----------------------------

// GL quadrature weights w_j for bandwidth b (2b values)
__global__ void k_quadw(float* out, int b) {
  int j = blockIdx.x * blockDim.x + threadIdx.x;
  if (j >= 2 * b) return;
  double t = PI_D * (2 * j + 1) / (4.0 * b);
  double s = 0.0;
  for (int k = 0; k < b; k++) s += sin(t * (2 * k + 1)) / (2 * k + 1);
  out[j] = (float)((2.0 / b) * sin(t) * s);
}

// Wigner small-d d^l_{m',m}(beta) via factorial-sum formula (double).
// Table layout: for l=0..lmax-1 blocks of [nbeta][2l+1][2l+1].
// bIn>0: beta_j = pi*(2j+1)/(4*bIn);  bIn==0: beta = pi/16 (kernel grid).
__global__ void k_wigner(float* out, int nbeta, int bIn, int lmax, long long total) {
  long long idx = blockIdx.x * (long long)blockDim.x + threadIdx.x;
  if (idx >= total) return;
  int l = 0; long long cum = 0;
  for (;;) {
    long long d = 2 * l + 1, sz = (long long)nbeta * d * d;
    if (idx < cum + sz) break;
    cum += sz; l++;
  }
  const int d = 2 * l + 1;
  long long rem = idx - cum;
  int j  = (int)(rem / ((long long)d * d));
  int r2 = (int)(rem % ((long long)d * d));
  int a = r2 / d, c = r2 % d;
  int mp = a - l, m = c - l;
  double beta = bIn ? PI_D * (2 * j + 1) / (4.0 * bIn) : PI_D / 16.0;
  double lf[64];
  lf[0] = 0.0;
  int need = 2 * l; if (need < 1) need = 1; if (need > 63) need = 63;
  for (int i = 1; i <= need; i++) lf[i] = lf[i - 1] + log((double)i);
  double ch = cos(0.5 * beta), sh = sin(0.5 * beta);
  int kmin = m - mp; if (kmin < 0) kmin = 0;
  int kmax = l + m; if (l - mp < kmax) kmax = l - mp;
  double pre = 0.5 * (lf[l + mp] + lf[l - mp] + lf[l + m] + lf[l - m]);
  double sum = 0.0;
  for (int k = kmin; k <= kmax; k++) {
    double lt = pre - lf[l + m - k] - lf[k] - lf[mp - m + k] - lf[l - mp - k];
    double t = exp(lt) * pow(ch, (double)(2 * l + m - mp - 2 * k)) *
               pow(sh, (double)(mp - m + 2 * k));
    sum += ((mp - m + k) & 1) ? -t : t;
  }
  out[idx] = (float)sum;
}

// Truncated DFT matrices, real-expanded, stored TRANSPOSED [Nf][Kf]:
// mode 0: real-in fwd   Bt[2Mf][N]
// mode 1: cplx-in fwd   Bt[2Mf][2N]
// mode 2: cplx inv      Bt[2N][2Mf]
// mode 3: real-out inv  Bt[N][2Mf]
__global__ void k_dftmat(float* out, int N, int Mf, int mode) {
  long long idx = blockIdx.x * (long long)blockDim.x + threadIdx.x;
  if (idx >= (long long)N * Mf) return;
  int p = (int)(idx % Mf), xn = (int)(idx / Mf);
  int L = (Mf - 1) / 2;
  double th = 2.0 * PI_D * (double)(p - L) * xn / N;
  float c = (float)cos(th), s = (float)sin(th);
  if (mode == 0) {
    out[(2 * p) * (long long)N + xn] = c;
    out[(2 * p + 1) * (long long)N + xn] = -s;
  } else if (mode == 1) {
    long long ld = 2 * N;
    out[(2 * p) * ld + 2 * xn] = c;      out[(2 * p) * ld + 2 * xn + 1] = s;
    out[(2 * p + 1) * ld + 2 * xn] = -s; out[(2 * p + 1) * ld + 2 * xn + 1] = c;
  } else if (mode == 2) {
    long long ld = 2 * Mf;
    out[(2 * xn) * ld + 2 * p] = c;      out[(2 * xn) * ld + 2 * p + 1] = -s;
    out[(2 * xn + 1) * ld + 2 * p] = s;  out[(2 * xn + 1) * ld + 2 * p + 1] = c;
  } else {
    long long ld = 2 * Mf;
    out[xn * ld + 2 * p] = c;
    out[xn * ld + 2 * p + 1] = -s;
  }
}

// S2 kernel FT -> per-l TRANSPOSED expanded B: Bt [2*fo*(2l+1)][2]
__global__ void k_kftS2(const float* kw, const float* TK, float* K2,
                        int fo, int bOut, long long total) {
  long long idx = blockIdx.x * (long long)blockDim.x + threadIdx.x;
  if (idx >= total) return;
  int l = 0; long long cum = 0, tko = 0, k2o = 0;
  for (;;) {
    long long d = 2 * l + 1, sz = (long long)fo * d;
    if (idx < cum + sz) break;
    cum += sz; tko += d * d; k2o += 4ll * fo * d; l++;
  }
  const int d = 2 * l + 1;
  long long t = idx - cum;
  int n = (int)(t % d), o = (int)(t / d);
  float dv = TK[tko + (long long)n * d + l];  // d^l_{n-l, 0}(pi/16)
  double cr = 0, ci = 0;
  for (int p = 0; p < 6; p++) {
    double al = 2.0 * PI_D * p / 6.0;
    double ph = al * (n - l);
    double kv = (double)kw[o * 6 + p] * (double)dv;
    cr += kv * cos(ph); ci -= kv * sin(ph);
  }
  long long c0 = 2ll * ((long long)o * d + n);
  float* B = K2 + k2o;
  B[c0 * 2 + 0] = (float)cr;        B[(c0 + 1) * 2 + 0] = (float)ci;
  B[c0 * 2 + 1] = (float)(-ci);     B[(c0 + 1) * 2 + 1] = (float)cr;
}

// SO3 kernel FT -> per-l TRANSPOSED expanded B: Bt [2*fo*(2l+1)][2*fi*(2l+1)]
// logical ((i,k);(o,n)) = conj( sum_p kw[i,o,p] d^l_{n,k}(pi/16)
//                               e^{i a_p (n-l)} e^{i g_p (k-l)} )
__global__ void k_kftSO3(const float* kw, const float* TK, float* K2,
                         int fi, int fo, int bOut, long long total) {
  long long idx = blockIdx.x * (long long)blockDim.x + threadIdx.x;
  if (idx >= total) return;
  int l = 0; long long cum = 0, tko = 0, k2o = 0;
  for (;;) {
    long long d = 2 * l + 1, sz = d * d * fi * fo;
    if (idx < cum + sz) break;
    cum += sz; tko += d * d; k2o += 4ll * fi * fo * d * d; l++;
  }
  const int d = 2 * l + 1;
  long long t = idx - cum;
  int n = (int)(t % d); t /= d;
  int o = (int)(t % fo); t /= fo;
  int k = (int)(t % d); t /= d;
  int i = (int)t;
  float dv = TK[tko + (long long)n * d + k];  // d^l_{n-l, k-l}(pi/16)
  double cr = 0, ci = 0;
  for (int p = 0; p < 36; p++) {
    double al = 2.0 * PI_D * (p / 6) / 6.0;
    double gm = 2.0 * PI_D * (p % 6) / 6.0;
    double ph = al * (n - l) + gm * (k - l);
    double kv = (double)kw[((long long)i * fo + o) * 36 + p] * (double)dv;
    cr += kv * cos(ph); ci -= kv * sin(ph);
  }
  long long ldK = 2ll * fi * d;
  long long r0 = 2ll * ((long long)i * d + k);
  long long c0 = 2ll * ((long long)o * d + n);
  float* B = K2 + k2o;
  B[c0 * ldK + r0] = (float)cr;           B[(c0 + 1) * ldK + r0] = (float)ci;
  B[c0 * ldK + r0 + 1] = (float)(-ci);    B[(c0 + 1) * ldK + r0 + 1] = (float)cr;
}

// MLP weight transpose: wt[n][k] = w[k][n]
__global__ void k_transp(const float* w, float* wt, int K, int N) {
  int idx = blockIdx.x * blockDim.x + threadIdx.x;
  if (idx >= K * N) return;
  int n = idx % N, k = idx / N;
  wt[(long long)n * K + k] = w[idx];
}

// -------------------------- pipeline VALU kernels --------------------------

// S2 Wigner contraction: xh[l][b][m] = sum_j w_j d^l_{m,0}(b_j) G[b,j,m+L]
__global__ void k_s2contract(const float* G, const float* T, const float* wq,
                             float* xh, int Bz, int bOut, int Nj, long long total) {
  long long idx = blockIdx.x * (long long)blockDim.x + threadIdx.x;
  if (idx >= total) return;
  const int L = bOut - 1, Mf = 2 * L + 1;
  long long per = 0; { long long s = 0; for (int l = 0; l < bOut; l++) s += 2 * l + 1; per = s; }
  long long z = idx / per, rem = idx % per;
  int l = 0; long long cum = 0, to = 0;
  for (;;) {
    long long d = 2 * l + 1;
    if (rem < cum + d) break;
    cum += d; to += (long long)Nj * d * d; l++;
  }
  const int d = 2 * l + 1;
  int m = (int)(rem - cum);
  double sr = 0, si = 0;
  for (int j = 0; j < Nj; j++) {
    float dv = T[to + ((long long)(j * d + m)) * d + l] * wq[j];
    const float* gp = G + ((z * Nj + j) * (long long)Mf + (m - l + L)) * 2;
    sr += (double)dv * gp[0]; si += (double)dv * gp[1];
  }
  float* o = xh + ((long long)Bz * cum + z * d + m) * 2;
  o[0] = (float)sr; o[1] = (float)si;
}

// SO3 Wigner contraction, output packed [l][b][m][(i,k)] so the per-l GEMM
// contracts a float-contiguous axis:
// xh[l][b][m][i][n] = sum_j w_j d^l_{m,n}(b_j) G2[(b,i),j,m-l+L,n-l+L]
__global__ void k_wigcontract(const float* G2, const float* T, const float* wq,
                              float* xh, int Z, int bOut, int Nj, long long total) {
  long long idx = blockIdx.x * (long long)blockDim.x + threadIdx.x;
  if (idx >= total) return;
  const int L = bOut - 1, Mf = 2 * L + 1;
  long long per = total / Z;
  long long z = idx / per, rem = idx % per;
  int l = 0; long long cum = 0, to = 0;
  for (;;) {
    long long d = 2 * l + 1, sz = d * d;
    if (rem < cum + sz) break;
    cum += sz; to += (long long)Nj * sz; l++;
  }
  const int d = 2 * l + 1;
  long long r2 = rem - cum;
  int m = (int)(r2 / d), n = (int)(r2 % d);
  double sr = 0, si = 0;
  for (int j = 0; j < Nj; j++) {
    float dv = T[to + ((long long)(j * d + m)) * d + n] * wq[j];
    const float* gp = G2 + (((z * Nj + j) * (long long)Mf + (m - l + L)) * Mf +
                            (n - l + L)) * 2;
    sr += (double)dv * gp[0]; si += (double)dv * gp[1];
  }
  const int fi = Z / 32;
  int bb = (int)(z / fi), ii = (int)(z % fi);
  float* o = xh + ((long long)Z * cum +
                   (((long long)bb * d + m) * fi + ii) * (long long)d + n) * 2;
  o[0] = (float)sr; o[1] = (float)si;
}

// Inverse assembly: Xf[z,j,p,q] = sum_{l>=lmin} (2l+1) d^l_{p-L,q-L}(b_j) yh_l[z,..]
__global__ void k_assemble(const float* yh, const float* T, float* Xf,
                           int Z, int bOut, long long total) {
  long long idx = blockIdx.x * (long long)blockDim.x + threadIdx.x;
  if (idx >= total) return;
  const int L = bOut - 1, Mf = 2 * L + 1, Nj = 2 * bOut;
  int q = (int)(idx % Mf); long long t = idx / Mf;
  int p = (int)(t % Mf); t /= Mf;
  int j = (int)(t % Nj); long long z = t / Nj;
  int mm = p - L, nn = q - L;
  int am = mm < 0 ? -mm : mm, an = nn < 0 ? -nn : nn;
  int lmin = am > an ? am : an;
  double sr = 0, si = 0;
  long long to = 0, yo = 0;
  for (int l = 0; l < bOut; l++) {
    long long d = 2 * l + 1;
    if (l >= lmin) {
      float dv = (float)(2 * l + 1) *
                 T[to + ((long long)(j * d + (mm + l))) * d + (nn + l)];
      const float* yp = yh + (yo + (z * d * d + (long long)(mm + l) * d + (nn + l))) * 2;
      sr += (double)dv * yp[0]; si += (double)dv * yp[1];
    }
    to += (long long)Nj * d * d;
    yo += (long long)Z * d * d;
  }
  float* o = Xf + idx * 2;
  o[0] = (float)sr; o[1] = (float)si;
}

// SO3 integrate: out[b,f] = sum_j w_j sum_{a,g} h[b,f,j,a,g] / 36   (b=3)
__global__ void k_integrate(const float* h, const float* wq, float* out, int BF) {
  int idx = blockIdx.x * blockDim.x + threadIdx.x;
  if (idx >= BF) return;
  const float* p = h + (long long)idx * 216;
  float s = 0.f;
  for (int j = 0; j < 6; j++) {
    float t = 0.f;
    for (int u = 0; u < 36; u++) t += p[j * 36 + u];
    s += wq[j] * t;
  }
  out[idx] = s / 36.f;
}

// Batchnorm over batch dim (32 rows)
__global__ void k_batchnorm(const float* x, const float* g, const float* be,
                            float* y, int F) {
  int f = blockIdx.x * blockDim.x + threadIdx.x;
  if (f >= F) return;
  float mu = 0.f;
  for (int r = 0; r < 32; r++) mu += x[r * F + f];
  mu *= (1.f / 32.f);
  float v = 0.f;
  for (int r = 0; r < 32; r++) { float d = x[r * F + f] - mu; v += d * d; }
  v *= (1.f / 32.f);
  float sc = rsqrtf(v + 1e-5f) * g[f];
  for (int r = 0; r < 32; r++) y[r * F + f] = (x[r * F + f] - mu) * sc + be[f];
}

}  // namespace

// ---------------------------------------------------------------------------
extern "C" void kernel_launch(void* const* d_in, const int* in_sizes, int n_in,
                              void* d_out, int out_size, void* d_ws, size_t ws_size,
                              hipStream_t stream) {
  (void)in_sizes; (void)n_in; (void)out_size; (void)ws_size;
  const float* x   = (const float*)d_in[0];
  const float* kS2 = (const float*)d_in[1];
  const float* kc[4] = {(const float*)d_in[2], (const float*)d_in[3],
                        (const float*)d_in[4], (const float*)d_in[5]};
  const float* g1 = (const float*)d_in[6];  const float* be1 = (const float*)d_in[7];
  const float* w1 = (const float*)d_in[8];  const float* b1  = (const float*)d_in[9];
  const float* g2 = (const float*)d_in[10]; const float* be2 = (const float*)d_in[11];
  const float* w2 = (const float*)d_in[12]; const float* b2  = (const float*)d_in[13];
  const float* g3 = (const float*)d_in[14]; const float* be3 = (const float*)d_in[15];
  const float* w3 = (const float*)d_in[16]; const float* b3  = (const float*)d_in[17];

  float* W = (float*)d_ws;
  long long off = 0;
  auto alloc = [&](long long n) { long long r = off; off += (n + 3) & ~3ll; return r; };
  auto sumsq = [](int L) { long long s = 0; for (int l = 0; l < L; l++) { long long d = 2 * l + 1; s += d * d; } return s; };

  const int NB = 5;
  const int QB[NB] = {30, 17, 9, 5, 3};
  auto idxOf = [&](int b) { for (int i = 0; i < NB; i++) if (QB[i] == b) return i; return 0; };

  long long qoff[NB], toff[NB];
  for (int i = 0; i < NB; i++) qoff[i] = alloc(2 * QB[i]);
  for (int i = 0; i < NB; i++) toff[i] = alloc((long long)2 * QB[i] * sumsq(QB[i]));
  long long tko = alloc(sumsq(30));

  struct LSpec { int fi, fo, bi, bo; };
  const LSpec LS[4] = {{2, 5, 30, 17}, {5, 12, 17, 9}, {12, 28, 9, 5}, {28, 64, 5, 3}};

  long long dS2 = alloc(60ll * 118);
  long long dF0[4], dF1[4];
  for (int i = 0; i < 4; i++) {
    int N = 2 * LS[i].bi, Mf = 2 * LS[i].bo - 1;
    dF0[i] = alloc((long long)N * 2 * Mf);
    dF1[i] = alloc(2ll * N * 2 * Mf);
  }
  long long dI2[NB], dI3[NB];
  for (int i = 0; i < NB; i++) {
    int N = 2 * QB[i], Mf = 2 * QB[i] - 1;
    dI2[i] = alloc(2ll * Mf * 2 * N);
    dI3[i] = alloc(2ll * Mf * N);
  }
  long long kS2o = alloc(8ll * 900);
  long long k2o[4];
  for (int i = 0; i < 4; i++) k2o[i] = alloc(4ll * LS[i].fi * LS[i].fo * sumsq(LS[i].bo));
  long long wT1 = alloc(64 * 64), wT2 = alloc(64 * 32), wT3 = alloc(32 * 10);

  // activation arenas (ping-pong): S spatial real, A/B spectral complex, X/Y packed
  long long maxS = 64ll * 60 * 60 * 60, maxA = 0, maxB = 0, maxX = 0, maxY = 0;
  { long long v;
    v = 1920ll * 59 * 2;            if (v > maxA) maxA = v;
    v = 64ll * 60 * 59 * 59 * 2;    if (v > maxA) maxA = v;
    v = 64ll * 60 * 59 * 60 * 2;    if (v > maxB) maxB = v;
    v = 32ll * 900 * 2;             if (v > maxX) maxX = v;
    v = 64ll * sumsq(30) * 2;       if (v > maxY) maxY = v; }
  for (int i = 0; i < 4; i++) {
    long long fi = LS[i].fi, fo = LS[i].fo;
    long long Nj = 2 * LS[i].bi, Mf = 2 * LS[i].bo - 1, NjO = 2 * LS[i].bo;
    long long Z = 32 * fi, Zo = 32 * fo, v;
    v = Z * Nj * Mf * Nj * 2;       if (v > maxA) maxA = v;
    v = Zo * NjO * Mf * Mf * 2;     if (v > maxA) maxA = v;
    v = Z * Nj * Mf * Mf * 2;       if (v > maxB) maxB = v;
    v = Zo * NjO * NjO * Mf * 2;    if (v > maxB) maxB = v;
    v = Z * sumsq(LS[i].bo) * 2;    if (v > maxX) maxX = v;
    v = Zo * sumsq(LS[i].bo) * 2;   if (v > maxY) maxY = v;
    v = Zo * NjO * NjO * NjO;       if (v > maxS) maxS = v;
  }
  long long bufS = alloc(maxS), bufA = alloc(maxA), bufB = alloc(maxB);
  long long bufX = alloc(maxX), bufY = alloc(maxY);
  long long m0 = alloc(32 * 64), m1 = alloc(32 * 64), m2 = alloc(32 * 64);

  auto g1d = [](long long n) { return dim3((unsigned)((n + 255) / 256)); };
  const int BIG = 1 << 30;
  auto GEMMF = [&](const GemmFArgs& ga, int Z) {
    dim3 grid((unsigned)((ga.M + 31) / 32), (unsigned)((ga.Nf + 15) / 16), (unsigned)Z);
    k_gemmf<<<grid, dim3(32), 0, stream>>>(ga);
  };

  // ---- constant-table generation (on device, every call: graph-safe) ----
  for (int i = 0; i < NB; i++)
    k_quadw<<<dim3(1), dim3(64), 0, stream>>>(W + qoff[i], QB[i]);
  for (int i = 0; i < NB; i++) {
    long long tot = 2ll * QB[i] * sumsq(QB[i]);
    k_wigner<<<g1d(tot), dim3(256), 0, stream>>>(W + toff[i], 2 * QB[i], QB[i], QB[i], tot);
  }
  { long long tot = sumsq(30);
    k_wigner<<<g1d(tot), dim3(256), 0, stream>>>(W + tko, 1, 0, 30, tot); }
  k_dftmat<<<g1d(60ll * 59), dim3(256), 0, stream>>>(W + dS2, 60, 59, 0);
  for (int i = 0; i < 4; i++) {
    int N = 2 * LS[i].bi, Mf = 2 * LS[i].bo - 1;
    k_dftmat<<<g1d((long long)N * Mf), dim3(256), 0, stream>>>(W + dF0[i], N, Mf, 0);
    k_dftmat<<<g1d((long long)N * Mf), dim3(256), 0, stream>>>(W + dF1[i], N, Mf, 1);
  }
  for (int i = 0; i < NB; i++) {
    int N = 2 * QB[i], Mf = 2 * QB[i] - 1;
    k_dftmat<<<g1d((long long)N * Mf), dim3(256), 0, stream>>>(W + dI2[i], N, Mf, 2);
    k_dftmat<<<g1d((long long)N * Mf), dim3(256), 0, stream>>>(W + dI3[i], N, Mf, 3);
  }
  { long long tot = 2ll * 900;
    k_kftS2<<<g1d(tot), dim3(256), 0, stream>>>(kS2, W + tko, W + kS2o, 2, 30, tot); }
  for (int i = 0; i < 4; i++) {
    long long tot = (long long)LS[i].fi * LS[i].fo * sumsq(LS[i].bo);
    k_kftSO3<<<g1d(tot), dim3(256), 0, stream>>>(kc[i], W + tko, W + k2o[i],
                                                 LS[i].fi, LS[i].fo, LS[i].bo, tot);
  }
  k_transp<<<g1d(64 * 64), dim3(256), 0, stream>>>(w1, W + wT1, 64, 64);
  k_transp<<<g1d(64 * 32), dim3(256), 0, stream>>>(w2, W + wT2, 64, 32);
  k_transp<<<g1d(32 * 10), dim3(256), 0, stream>>>(w3, W + wT3, 32, 10);

  // ---- Layer 0: S2 conv (b=30) -> relu -> spatial [32,2,60,60,60] ----
  {
    const int bo = 30, Mf = 59, Nj = 60;
    const int ti = idxOf(30);
    { GemmFArgs ga{};  // alpha DFT of x: rows (b,j), contiguous K=a(60)
      ga.A = x; ga.saz = 0; ga.sam = 60; ga.M = 32 * 60; ga.Kf = 60;
      ga.Bt = W + dS2; ga.ldbt = 60; ga.Nf = 2 * Mf;
      ga.C = W + bufA; ga.scz = 0; ga.scm = 2 * Mf; ga.sco = 0; ga.scn = 2; ga.ndiv = BIG;
      ga.cCplx = 1; ga.bias = nullptr; ga.relu = 0;
      GEMMF(ga, 1); }
    { long long tot = 32ll * 900;
      k_s2contract<<<g1d(tot), dim3(256), 0, stream>>>(W + bufA, W + toff[ti],
                                                       W + qoff[ti], W + bufX, 32, bo, Nj, tot); }
    { long long xo = bufX, yo = bufY, ko = kS2o;  // per-l kernel mixing
      for (int l = 0; l < bo; l++) {
        int d = 2 * l + 1;
        GemmFArgs gc{};
        gc.A = W + xo; gc.saz = (long long)d * 2; gc.sam = 2; gc.M = d; gc.Kf = 2;
        gc.Bt = W + ko; gc.ldbt = 2; gc.Nf = 4 * d;
        gc.C = W + yo; gc.scz = 2ll * d * d * 2; gc.scm = (long long)d * 2;
        gc.sco = (long long)d * d * 2; gc.scn = 2; gc.ndiv = d; gc.cCplx = 1;
        gc.bias = nullptr; gc.relu = 0;
        GEMMF(gc, 32);
        xo += 32ll * d * 2; yo += 64ll * d * d * 2; ko += 8ll * d;
      } }
    const long long ZO = 64;
    { long long tot = ZO * 60 * 59 * 59;
      k_assemble<<<g1d(tot), dim3(256), 0, stream>>>(W + bufY, W + toff[ti], W + bufA,
                                                     (int)ZO, bo, tot); }
    { GemmFArgs ga{};  // inverse gamma DFT: z=(zo,j), rows p, contig K=q; out [z][g][p]
      ga.A = W + bufA; ga.saz = (long long)Mf * Mf * 2; ga.sam = (long long)Mf * 2;
      ga.M = Mf; ga.Kf = 2 * Mf;
      ga.Bt = W + dI2[ti]; ga.ldbt = 2 * Mf; ga.Nf = 2 * 60;
      ga.C = W + bufB; ga.scz = 60ll * Mf * 2; ga.scm = 2; ga.sco = 0;
      ga.scn = (long long)Mf * 2; ga.ndiv = BIG; ga.cCplx = 1; ga.bias = nullptr; ga.relu = 0;
      GEMMF(ga, (int)(ZO * 60)); }
    { GemmFArgs ga{};  // inverse alpha DFT + Re + ReLU: rows g, contig K=p
      ga.A = W + bufB; ga.saz = 60ll * Mf * 2; ga.sam = (long long)Mf * 2;
      ga.M = 60; ga.Kf = 2 * Mf;
      ga.Bt = W + dI3[ti]; ga.ldbt = 2 * Mf; ga.Nf = 60;
      ga.C = W + bufS; ga.scz = 60ll * 60; ga.scm = 1; ga.sco = 0; ga.scn = 60; ga.ndiv = BIG;
      ga.cCplx = 0; ga.bias = nullptr; ga.relu = 1;
      GEMMF(ga, (int)(ZO * 60)); }
  }

  // ---- Layers 1..4: SO3 convs ----
  for (int i = 0; i < 4; i++) {
    const int fi = LS[i].fi, fo = LS[i].fo, bi = LS[i].bi, bo = LS[i].bo;
    const int Nj = 2 * bi, Ng = 2 * bi, Mf = 2 * bo - 1;
    const int NjO = 2 * bo, NgO = 2 * bo;
    const int Z = 32 * fi, ZO = 32 * fo;
    const int tbi = idxOf(bi), tbo = idxOf(bo);
    { GemmFArgs ga{};  // fwd gamma DFT: z=(b,i,j), rows a, contig K=g; out [z][q][a]
      ga.A = W + bufS; ga.saz = (long long)Ng * Ng; ga.sam = Ng; ga.M = Ng; ga.Kf = Ng;
      ga.Bt = W + dF0[i]; ga.ldbt = Ng; ga.Nf = 2 * Mf;
      ga.C = W + bufA; ga.scz = (long long)Mf * Ng * 2; ga.scm = 2; ga.sco = 0;
      ga.scn = (long long)Ng * 2; ga.ndiv = BIG; ga.cCplx = 1; ga.bias = nullptr; ga.relu = 0;
      GEMMF(ga, Z * Nj); }
    { GemmFArgs ga{};  // fwd alpha DFT: z=(b,i,j), rows q, contig K=a; out [z][p][q]
      ga.A = W + bufA; ga.saz = (long long)Mf * Ng * 2; ga.sam = (long long)Ng * 2;
      ga.M = Mf; ga.Kf = 2 * Ng;
      ga.Bt = W + dF1[i]; ga.ldbt = 2 * Ng; ga.Nf = 2 * Mf;
      ga.C = W + bufB; ga.scz = (long long)Mf * Mf * 2; ga.scm = 2; ga.sco = 0;
      ga.scn = (long long)Mf * 2; ga.ndiv = BIG; ga.cCplx = 1; ga.bias = nullptr; ga.relu = 0;
      GEMMF(ga, Z * Nj); }
    { long long tot = (long long)Z * sumsq(bo);
      k_wigcontract<<<g1d(tot), dim3(256), 0, stream>>>(W + bufB, W + toff[tbi],
                                                        W + qoff[tbi], W + bufX, Z, bo, Nj, tot); }
    { long long xo = bufX, yo = bufY, ko = k2o[i];  // per-l channel mixing
      for (int l = 0; l < bo; l++) {
        int d = 2 * l + 1;
        GemmFArgs gc{};
        gc.A = W + xo; gc.saz = (long long)d * fi * d * 2; gc.sam = (long long)fi * d * 2;
        gc.M = d; gc.Kf = 2 * fi * d;
        gc.Bt = W + ko; gc.ldbt = 2 * fi * d; gc.Nf = 2 * fo * d;
        gc.C = W + yo; gc.scz = (long long)fo * d * d * 2; gc.scm = (long long)d * 2;
        gc.sco = (long long)d * d * 2; gc.scn = 2; gc.ndiv = d; gc.cCplx = 1;
        gc.bias = nullptr; gc.relu = 0;
        GEMMF(gc, 32);
        xo += (long long)Z * d * d * 2; yo += (long long)ZO * d * d * 2;
        ko += 4ll * fi * fo * d * d;
      } }
    { long long tot = (long long)ZO * NjO * Mf * Mf;
      k_assemble<<<g1d(tot), dim3(256), 0, stream>>>(W + bufY, W + toff[tbo], W + bufA,
                                                     ZO, bo, tot); }
    { GemmFArgs ga{};  // inverse gamma DFT: z=(zo,j), rows p, contig K=q; out [z][g][p]
      ga.A = W + bufA; ga.saz = (long long)Mf * Mf * 2; ga.sam = (long long)Mf * 2;
      ga.M = Mf; ga.Kf = 2 * Mf;
      ga.Bt = W + dI2[tbo]; ga.ldbt = 2 * Mf; ga.Nf = 2 * NgO;
      ga.C = W + bufB; ga.scz = (long long)NgO * Mf * 2; ga.scm = 2; ga.sco = 0;
      ga.scn = (long long)Mf * 2; ga.ndiv = BIG; ga.cCplx = 1; ga.bias = nullptr; ga.relu = 0;
      GEMMF(ga, ZO * NjO); }
    { GemmFArgs ga{};  // inverse alpha DFT + Re + ReLU: rows g, contig K=p
      ga.A = W + bufB; ga.saz = (long long)NgO * Mf * 2; ga.sam = (long long)Mf * 2;
      ga.M = NgO; ga.Kf = 2 * Mf;
      ga.Bt = W + dI3[tbo]; ga.ldbt = 2 * Mf; ga.Nf = NgO;
      ga.C = W + bufS; ga.scz = (long long)NgO * NgO; ga.scm = 1; ga.sco = 0; ga.scn = NgO;
      ga.ndiv = BIG; ga.cCplx = 0; ga.bias = nullptr; ga.relu = 1;
      GEMMF(ga, ZO * NjO); }
  }

  // ---- integrate + MLP head ----
  k_integrate<<<g1d(2048), dim3(256), 0, stream>>>(W + bufS, W + qoff[idxOf(3)], W + m0, 2048);
  k_batchnorm<<<dim3(1), dim3(64), 0, stream>>>(W + m0, g1, be1, W + m1, 64);
  { GemmFArgs ga{};
    ga.A = W + m1; ga.saz = 0; ga.sam = 64; ga.M = 32; ga.Kf = 64;
    ga.Bt = W + wT1; ga.ldbt = 64; ga.Nf = 64;
    ga.C = W + m2; ga.scz = 0; ga.scm = 64; ga.sco = 0; ga.scn = 1; ga.ndiv = BIG; ga.cCplx = 0;
    ga.bias = b1; ga.relu = 1;
    GEMMF(ga, 1); }
  k_batchnorm<<<dim3(1), dim3(64), 0, stream>>>(W + m2, g2, be2, W + m1, 64);
  { GemmFArgs ga{};
    ga.A = W + m1; ga.saz = 0; ga.sam = 64; ga.M = 32; ga.Kf = 64;
    ga.Bt = W + wT2; ga.ldbt = 64; ga.Nf = 32;
    ga.C = W + m0; ga.scz = 0; ga.scm = 32; ga.sco = 0; ga.scn = 1; ga.ndiv = BIG; ga.cCplx = 0;
    ga.bias = b2; ga.relu = 1;
    GEMMF(ga, 1); }
  k_batchnorm<<<dim3(1), dim3(32), 0, stream>>>(W + m0, g3, be3, W + m1, 32);
  { GemmFArgs ga{};
    ga.A = W + m1; ga.saz = 0; ga.sam = 32; ga.M = 32; ga.Kf = 32;
    ga.Bt = W + wT3; ga.ldbt = 32; ga.Nf = 10;
    ga.C = (float*)d_out; ga.scz = 0; ga.scm = 10; ga.sco = 0; ga.scn = 1; ga.ndiv = BIG;
    ga.cCplx = 0; ga.bias = b3; ga.relu = 0;
    GEMMF(ga, 1); }
}
